// STTransformerBlock_81303730913778
// MI455X (gfx1250) — matrix operand order repaired and checked by reading, not compiled, not run on hardware
//
#include <hip/hip_runtime.h>
#include <hip/hip_bf16.h>

// ------------------------------------------------------------------
// Types
// ------------------------------------------------------------------
typedef _Float16 h8  __attribute__((ext_vector_type(8)));
typedef _Float16 h16 __attribute__((ext_vector_type(16)));
typedef float    f8v __attribute__((ext_vector_type(8)));

static constexpr int cB = 4, cT = 16, cN = 256, cC = 768, cH = 12, cD = 64, cHID = 3072;
static constexpr int TOK = cB * cT * cN;          // 16384
static constexpr float SCALE = 0.125f;            // D^-0.5

// ------------------------------------------------------------------
// Fragment helpers (gfx1250 wave32 WMMA 16x16x32 f16 layouts)
// A (16x32, MxK): lane m=l%16, kh=l/16. lo8 -> K = kh*8..+7, hi8 -> K = 16+kh*8..+7
// B (32x16, KxN) stored transposed BT[n][k]: lane n=l%16, kh=l/16. lo8 -> K = kh*8..+7, hi8 -> K = 16+kh*8..+7
// C/D (16x16): lane n=l%16, reg r -> m = 8*(l/16)+r
// ------------------------------------------------------------------
__device__ __forceinline__ h16 cat8(h8 a, h8 b) {
  return __builtin_shufflevector(a, b, 0,1,2,3,4,5,6,7,8,9,10,11,12,13,14,15);
}
__device__ __forceinline__ h16 load_frag_a(const _Float16* rowp, int kh) {
  h8 lo = *(const h8*)(rowp + kh * 8);
  h8 hi = *(const h8*)(rowp + 16 + kh * 8);
  return cat8(lo, hi);
}
__device__ __forceinline__ h16 load_frag_b(const _Float16* rowp, int kh) {
  h8 lo = *(const h8*)(rowp + kh * 8);
  h8 hi = *(const h8*)(rowp + 16 + kh * 8);
  return cat8(lo, hi);
}
__device__ __forceinline__ f8v wmma32(h16 a, h16 b, f8v c) {
  return __builtin_amdgcn_wmma_f32_16x16x32_f16(false, a, false, b, (short)0, c, false, false);
}
__device__ __forceinline__ float rmax16(float v) {
  v = fmaxf(v, __shfl_xor(v, 1, 16));
  v = fmaxf(v, __shfl_xor(v, 2, 16));
  v = fmaxf(v, __shfl_xor(v, 4, 16));
  v = fmaxf(v, __shfl_xor(v, 8, 16));
  return v;
}
__device__ __forceinline__ float rsum16(float v) {
  v += __shfl_xor(v, 1, 16);
  v += __shfl_xor(v, 2, 16);
  v += __shfl_xor(v, 4, 16);
  v += __shfl_xor(v, 8, 16);
  return v;
}

// ------------------------------------------------------------------
// CDNA5 async global -> LDS copy (16B per lane) + wait.
// Low 32 bits of a generic pointer to __shared__ are the LDS byte
// offset, which is what the VDST operand of the async op takes.
// ------------------------------------------------------------------
__device__ __forceinline__ void async_copy_b128(void* lds_dst, const void* gsrc) {
  unsigned ldsoff = (unsigned)(uintptr_t)lds_dst;
  asm volatile("global_load_async_to_lds_b128 %0, %1, off"
               :: "v"(ldsoff), "v"(gsrc)
               : "memory");
}
__device__ __forceinline__ void async_wait0() {
  asm volatile("s_wait_asynccnt 0x0" ::: "memory");
}

// ------------------------------------------------------------------
// f32 -> f16 convert
// ------------------------------------------------------------------
__global__ void cvt_f32_f16(const float* __restrict__ x, _Float16* __restrict__ y, int n) {
  int i = blockIdx.x * 256 + threadIdx.x;
  if (i < n) y[i] = (_Float16)x[i];
}

// ------------------------------------------------------------------
// LayerNorm (C=768), f32 in -> f16 out
// ------------------------------------------------------------------
__global__ __launch_bounds__(256) void ln_f16(const float* __restrict__ X,
                                              const float* __restrict__ g,
                                              const float* __restrict__ bb,
                                              _Float16* __restrict__ Y) {
  __shared__ float sA[256], sB[256];
  const int row = blockIdx.x, t = threadIdx.x;
  const float* xp = X + (size_t)row * cC;
  float a0 = xp[t], a1 = xp[t + 256], a2 = xp[t + 512];
  sA[t] = a0 + a1 + a2;
  sB[t] = a0 * a0 + a1 * a1 + a2 * a2;
  __syncthreads();
  for (int off = 128; off > 0; off >>= 1) {
    if (t < off) { sA[t] += sA[t + off]; sB[t] += sB[t + off]; }
    __syncthreads();
  }
  const float mean = sA[0] * (1.0f / cC);
  const float var  = sB[0] * (1.0f / cC) - mean * mean;
  const float rstd = rsqrtf(var + 1e-5f);
  _Float16* yp = Y + (size_t)row * cC;
  yp[t]       = (_Float16)((a0 - mean) * rstd * g[t]       + bb[t]);
  yp[t + 256] = (_Float16)((a1 - mean) * rstd * g[t + 256] + bb[t + 256]);
  yp[t + 512] = (_Float16)((a2 - mean) * rstd * g[t + 512] + bb[t + 512]);
}

// ------------------------------------------------------------------
// Generic f16 GEMM: C[M,N] = A[M,K] @ B[K,N] (+bias,+residual / GELU)
// WG: 256 threads, tile 64(M) x 128(N); wave tile 16 x 64; K step 32.
// Double-buffered: async global->LDS A staging + manual transposed B
// staging for step i+1 overlap the WMMA compute of step i.
// EPI: 0 = write f16 raw; 1 = f32 out = res + acc + bias; 2 = f16 gelu(acc+bias)
// ------------------------------------------------------------------
template <int EPI>
__global__ __launch_bounds__(256) void gemm_f16(const _Float16* __restrict__ A,
                                                const _Float16* __restrict__ Bw,
                                                const float* __restrict__ bias,
                                                const float* __restrict__ res,
                                                float* __restrict__ outf,
                                                _Float16* __restrict__ outh,
                                                int M, int Nn, int K) {
  constexpr int ASZ = 64 * 40;   // one A buffer (f16 elems)
  constexpr int BSZ = 128 * 40;  // one B buffer (f16 elems)
  __shared__ __align__(16) _Float16 As[2 * ASZ];
  __shared__ __align__(16) _Float16 Bs[2 * BSZ];
  const int t = threadIdx.x;
  const int n0 = blockIdx.x * 128;
  const int m0 = blockIdx.y * 64;
  const int wid = t >> 5, lane = t & 31;
  const int wm = wid & 3, wn = wid >> 2;
  const int kh = lane >> 4, cn = lane & 15;

  // staging coordinates (fixed per thread)
  const int ar = t >> 2, akc = (t & 3) * 8;  // A: row, col-group
  const int bkr = t & 31, bng = t >> 5;      // B: k row, col-group base

  auto stageA = [&](int buf, int k0) {
    async_copy_b128(As + buf * ASZ + ar * 40 + akc,
                    A + (size_t)(m0 + ar) * K + k0 + akc);
  };
  auto stageB = [&](int buf, int k0) {
    _Float16* bs = Bs + buf * BSZ;
#pragma unroll
    for (int rep = 0; rep < 2; ++rep) {
      int nc = bng + rep * 8;  // 0..15 column-groups of 8
      h8 bv = *(const h8*)(Bw + (size_t)(k0 + bkr) * Nn + n0 + nc * 8);
#pragma unroll
      for (int j = 0; j < 8; ++j) bs[(nc * 8 + j) * 40 + bkr] = bv[j];
    }
  };

  f8v acc[4] = {};
  const int niter = K >> 5;

  stageA(0, 0);
  stageB(0, 0);
  async_wait0();
  __syncthreads();

  for (int i = 0; i < niter; ++i) {
    const int cur = i & 1;
    const int k0n = (i + 1) << 5;
    if (i + 1 < niter) {
      stageA(1 - cur, k0n);  // async: overlaps with WMMA below
      stageB(1 - cur, k0n);
      if (i + 2 < niter) {  // prefetch 2 steps ahead (global_prefetch_b8)
        __builtin_prefetch(A + (size_t)(m0 + ar) * K + k0n + 32, 0, 0);
        __builtin_prefetch(Bw + (size_t)(k0n + 32 + bkr) * Nn + n0, 0, 0);
      }
    }
    // compute on current buffer
    h16 af = load_frag_a(As + cur * ASZ + (wm * 16 + cn) * 40, kh);
#pragma unroll
    for (int nt = 0; nt < 4; ++nt) {
      h16 bf = load_frag_b(Bs + cur * BSZ + (wn * 64 + nt * 16 + cn) * 40, kh);
      acc[nt] = wmma32(af, bf, acc[nt]);
    }
    if (i + 1 < niter) async_wait0();  // next-buffer async landed
    __syncthreads();
  }

  const int half = lane >> 4;
#pragma unroll
  for (int nt = 0; nt < 4; ++nt) {
    int col = n0 + wn * 64 + nt * 16 + cn;
    float bia = (EPI != 0) ? bias[col] : 0.0f;
#pragma unroll
    for (int r = 0; r < 8; ++r) {
      size_t idx = (size_t)(m0 + wm * 16 + half * 8 + r) * Nn + col;
      float v = acc[nt][r];
      if (EPI == 0) {
        outh[idx] = (_Float16)v;
      } else if (EPI == 1) {
        outf[idx] = res[idx] + v + bia;
      } else {
        float gg = v + bia;
        outh[idx] = (_Float16)(0.5f * gg * (1.0f + erff(gg * 0.70710678118f)));
      }
    }
  }
}

// ------------------------------------------------------------------
// Spatial attention: one WG per (b*t, h). 8 waves; each wave 2x 16-row
// Q tiles, flash-style online softmax over 256 keys in chunks of 32.
// qkv layout: [(b t n), 3*C] with q|k|v blocks of 768, head offset h*64.
// ------------------------------------------------------------------
__global__ __launch_bounds__(256) void spatial_attn(const _Float16* __restrict__ qkv,
                                                    _Float16* __restrict__ o) {
  __shared__ __align__(16) _Float16 Vt[64 * 264];     // V transposed: [d][key]
  __shared__ __align__(16) _Float16 Ps[8][16 * 32];   // per-wave P staging
  const int bt = blockIdx.x / cH, h = blockIdx.x % cH;
  const int t = threadIdx.x, wid = t >> 5, lane = t & 31;
  const int kh = lane >> 4, cn = lane & 15, half = kh;
  const size_t base = (size_t)bt * cN * (3 * cC);
  const int qoff = h * 64, koff = cC + h * 64, voff = 2 * cC + h * 64;

  {  // stage Vt
    int key = t;
    const _Float16* vp = qkv + base + (size_t)key * (3 * cC) + voff;
#pragma unroll
    for (int dg = 0; dg < 8; ++dg) {
      h8 v = *(const h8*)(vp + dg * 8);
#pragma unroll
      for (int j = 0; j < 8; ++j) Vt[(dg * 8 + j) * 264 + key] = v[j];
    }
  }
  __syncthreads();

  for (int rt = 0; rt < 2; ++rt) {
    const int rowbase = (wid * 2 + rt) * 16;
    h16 Aq[2];
    {
      const _Float16* qp = qkv + base + (size_t)(rowbase + cn) * (3 * cC) + qoff;
      Aq[0] = load_frag_a(qp, kh);       // d 0..31
      Aq[1] = load_frag_a(qp + 32, kh);  // d 32..63
    }
    f8v O[4] = {};
    float mprev[8], lsum[8];
#pragma unroll
    for (int r = 0; r < 8; ++r) { mprev[r] = -1e30f; lsum[r] = 0.0f; }

    for (int j0 = 0; j0 < cN; j0 += 32) {
      f8v s[2] = {};
#pragma unroll
      for (int nt = 0; nt < 2; ++nt) {
        const _Float16* kp = qkv + base + (size_t)(j0 + nt * 16 + cn) * (3 * cC) + koff;
        s[nt] = wmma32(Aq[0], load_frag_b(kp, kh), s[nt]);
        s[nt] = wmma32(Aq[1], load_frag_b(kp + 32, kh), s[nt]);
      }
      float p0[8], p1[8];
#pragma unroll
      for (int r = 0; r < 8; ++r) {
        float s0 = s[0][r] * SCALE;
        float s1 = s[1][r] * SCALE;
        float mc = rmax16(fmaxf(s0, s1));
        float mn = fmaxf(mprev[r], mc);
        float fac = __expf(mprev[r] - mn);
        p0[r] = __expf(s0 - mn);
        p1[r] = __expf(s1 - mn);
        float ps = rsum16(p0[r] + p1[r]);
        lsum[r] = lsum[r] * fac + ps;
        mprev[r] = mn;
        O[0][r] *= fac; O[1][r] *= fac; O[2][r] *= fac; O[3][r] *= fac;
        Ps[wid][(half * 8 + r) * 32 + cn]      = (_Float16)p0[r];
        Ps[wid][(half * 8 + r) * 32 + 16 + cn] = (_Float16)p1[r];
      }
      h16 Pa = load_frag_a(&Ps[wid][cn * 32], kh);
#pragma unroll
      for (int dt = 0; dt < 4; ++dt) {
        h16 bv = load_frag_b(Vt + (dt * 16 + cn) * 264 + j0, kh);
        O[dt] = wmma32(Pa, bv, O[dt]);
      }
    }
    const size_t obase = (size_t)bt * cN * cC;
#pragma unroll
    for (int dt = 0; dt < 4; ++dt) {
#pragma unroll
      for (int r = 0; r < 8; ++r) {
        int row = rowbase + half * 8 + r;
        float val = O[dt][r] / lsum[r];
        o[obase + (size_t)row * cC + h * 64 + dt * 16 + cn] = (_Float16)val;
      }
    }
  }
}

// ------------------------------------------------------------------
// Temporal causal attention: one wave per (b, n, h); T = 16.
// ------------------------------------------------------------------
__global__ __launch_bounds__(256) void temporal_attn(const _Float16* __restrict__ qkv,
                                                     _Float16* __restrict__ o) {
  __shared__ __align__(16) _Float16 Ps[8][16 * 32];
  __shared__ __align__(16) _Float16 Vt[8][64 * 32];
  const int t = threadIdx.x, wid = t >> 5, lane = t & 31;
  const int kh = lane >> 4, cn = lane & 15, half = kh;
  const int item = blockIdx.x * 8 + wid;
  const int b  = item / (cN * cH);
  const int rm = item % (cN * cH);
  const int n  = rm / cH;
  const int h  = rm % cH;
  const int qoff = h * 64, koff = cC + h * 64, voff = 2 * cC + h * 64;

  auto rowaddr = [&](int tt) -> size_t {
    return ((size_t)(b * cT + tt) * cN + n) * (3 * cC);
  };

  {  // stage zero-padded V^T per wave: Vt[d][key], key 16..31 zero
    int key = lane;
    for (int d = 0; d < 64; ++d) {
      _Float16 v = (_Float16)0.0f;
      if (key < 16) v = qkv[rowaddr(key) + voff + d];
      Vt[wid][d * 32 + key] = v;
    }
  }

  h16 Aq[2];
  {
    const _Float16* qp = qkv + rowaddr(cn) + qoff;
    Aq[0] = load_frag_a(qp, kh);
    Aq[1] = load_frag_a(qp + 32, kh);
  }
  f8v s = {};
  {
    const _Float16* kp = qkv + rowaddr(cn) + koff;  // key = cn
    s = wmma32(Aq[0], load_frag_b(kp, kh), s);
    s = wmma32(Aq[1], load_frag_b(kp + 32, kh), s);
  }
#pragma unroll
  for (int r = 0; r < 8; ++r) {
    int row = half * 8 + r;
    float sv = s[r] * SCALE;
    if (cn > row) sv = -1e30f;  // causal mask
    float mx = rmax16(sv);
    float e = __expf(sv - mx);
    float sum = rsum16(e);
    Ps[wid][row * 32 + cn]      = (_Float16)(e / sum);
    Ps[wid][row * 32 + 16 + cn] = (_Float16)0.0f;  // K padding
  }
  h16 Pa = load_frag_a(&Ps[wid][cn * 32], kh);
  f8v O[4] = {};
#pragma unroll
  for (int dt = 0; dt < 4; ++dt)
    O[dt] = wmma32(Pa, load_frag_b(&Vt[wid][(dt * 16 + cn) * 32], kh), O[dt]);
#pragma unroll
  for (int dt = 0; dt < 4; ++dt) {
#pragma unroll
    for (int r = 0; r < 8; ++r) {
      int rowt = half * 8 + r;
      o[((size_t)(b * cT + rowt) * cN + n) * cC + h * 64 + dt * 16 + cn] = (_Float16)O[dt][r];
    }
  }
}

// ------------------------------------------------------------------
// Launcher
// ------------------------------------------------------------------
extern "C" void kernel_launch(void* const* d_in, const int* in_sizes, int n_in,
                              void* d_out, int out_size, void* d_ws, size_t ws_size,
                              hipStream_t stream) {
  (void)in_sizes; (void)n_in; (void)out_size; (void)ws_size;

  const float* x       = (const float*)d_in[0];
  const float* ln1s_g  = (const float*)d_in[1];
  const float* ln1s_b  = (const float*)d_in[2];
  const float* ln1t_g  = (const float*)d_in[3];
  const float* ln1t_b  = (const float*)d_in[4];
  const float* ln2_g   = (const float*)d_in[5];
  const float* ln2_b   = (const float*)d_in[6];
  const float* Wqkv_s  = (const float*)d_in[7];
  const float* Wproj_s = (const float*)d_in[8];
  const float* bproj_s = (const float*)d_in[9];
  const float* Wqkv_t  = (const float*)d_in[10];
  const float* Wproj_t = (const float*)d_in[11];
  const float* bproj_t = (const float*)d_in[12];
  const float* Wfc1    = (const float*)d_in[13];
  const float* bfc1    = (const float*)d_in[14];
  const float* Wfc2    = (const float*)d_in[15];
  const float* bfc2    = (const float*)d_in[16];
  float* out = (float*)d_out;

  // --- workspace partition (256B aligned) ---
  char* wsb = (char*)d_ws;
  size_t off = 0;
  auto alloc = [&](size_t bytes) -> void* {
    off = (off + 255) & ~(size_t)255;
    void* p = wsb + off;
    off += bytes;
    return p;
  };
  _Float16* wqkv_s16  = (_Float16*)alloc((size_t)cC * 3 * cC * 2);
  _Float16* wproj_s16 = (_Float16*)alloc((size_t)cC * cC * 2);
  _Float16* wqkv_t16  = (_Float16*)alloc((size_t)cC * 3 * cC * 2);
  _Float16* wproj_t16 = (_Float16*)alloc((size_t)cC * cC * 2);
  _Float16* wfc1_16   = (_Float16*)alloc((size_t)cC * cHID * 2);
  _Float16* wfc2_16   = (_Float16*)alloc((size_t)cHID * cC * 2);
  _Float16* xh  = (_Float16*)alloc((size_t)TOK * cC * 2);        // LN output
  _Float16* big = (_Float16*)alloc((size_t)TOK * cHID * 2);      // qkv / MLP hidden
  _Float16* ao  = (_Float16*)alloc((size_t)TOK * cC * 2);        // attn output (pre-proj)
  float*    x1  = (float*)   alloc((size_t)TOK * cC * 4);        // after spatial residual
  float*    x2  = (float*)   alloc((size_t)TOK * cC * 4);        // after temporal residual

  // --- weight conversion f32 -> f16 ---
  auto cvt = [&](const float* src, _Float16* dst, int n) {
    cvt_f32_f16<<<(n + 255) / 256, 256, 0, stream>>>(src, dst, n);
  };
  cvt(Wqkv_s,  wqkv_s16,  cC * 3 * cC);
  cvt(Wproj_s, wproj_s16, cC * cC);
  cvt(Wqkv_t,  wqkv_t16,  cC * 3 * cC);
  cvt(Wproj_t, wproj_t16, cC * cC);
  cvt(Wfc1,    wfc1_16,   cC * cHID);
  cvt(Wfc2,    wfc2_16,   cHID * cC);

  const dim3 blk(256);

  // ---- sublayer 1: spatial attention ----
  ln_f16<<<TOK, blk, 0, stream>>>(x, ln1s_g, ln1s_b, xh);
  gemm_f16<0><<<dim3(3 * cC / 128, TOK / 64), blk, 0, stream>>>(
      xh, wqkv_s16, nullptr, nullptr, nullptr, big, TOK, 3 * cC, cC);
  spatial_attn<<<cB * cT * cH, blk, 0, stream>>>(big, ao);
  gemm_f16<1><<<dim3(cC / 128, TOK / 64), blk, 0, stream>>>(
      ao, wproj_s16, bproj_s, x, x1, nullptr, TOK, cC, cC);

  // ---- sublayer 2: temporal (causal) attention ----
  ln_f16<<<TOK, blk, 0, stream>>>(x1, ln1t_g, ln1t_b, xh);
  gemm_f16<0><<<dim3(3 * cC / 128, TOK / 64), blk, 0, stream>>>(
      xh, wqkv_t16, nullptr, nullptr, nullptr, big, TOK, 3 * cC, cC);
  temporal_attn<<<(cB * cN * cH) / 8, blk, 0, stream>>>(big, ao);
  gemm_f16<1><<<dim3(cC / 128, TOK / 64), blk, 0, stream>>>(
      ao, wproj_t16, bproj_t, x1, x2, nullptr, TOK, cC, cC);

  // ---- sublayer 3: MLP ----
  ln_f16<<<TOK, blk, 0, stream>>>(x2, ln2_g, ln2_b, xh);
  gemm_f16<2><<<dim3(cHID / 128, TOK / 64), blk, 0, stream>>>(
      xh, wfc1_16, bfc1, nullptr, nullptr, big, TOK, cHID, cC);
  gemm_f16<1><<<dim3(cC / 128, TOK / 64), blk, 0, stream>>>(
      big, wfc2_16, bfc2, x2, out, nullptr, TOK, cC, cHID);
}
